// TopKRouter_88811333746986
// MI455X (gfx1250) — matrix operand order repaired and checked by reading, not compiled, not run on hardware
//
#include <hip/hip_runtime.h>

typedef __attribute__((ext_vector_type(16))) __bf16 v16bf;
typedef __attribute__((ext_vector_type(8)))  float  v8f;
typedef __attribute__((ext_vector_type(4)))  float  v4f;
typedef unsigned int v4u __attribute__((ext_vector_type(4)));
typedef int          v8i __attribute__((ext_vector_type(8)));
typedef int          v4i __attribute__((ext_vector_type(4)));

#define NTOK 16384
#define HDIM 2048
#define NEXP 64
#define LROW 65            // padded LDS row (64 + 1) to avoid bank conflicts

// d_out layout (floats, concatenated in reference return order)
#define IDX_OFF   0        // top_indices  (N*2), stored as float(index)
#define WT_OFF    32768    // top_weights  (N*2)
#define LOSS_OFF  65536    // balance_loss (1)
#define FRAC_OFF  65537    // fraction_routed (64)
#define PROB_OFF  65601    // router_probs (N*64)

#define WP_ELEMS  (NEXP * HDIM)      // 131072 bf16 = 256 KB

// ---------------------------------------------------------------- zero ws accumulators
__global__ __launch_bounds__(128) void zero_ws_k(float* __restrict__ ws) {
    ws[threadIdx.x] = 0.0f;   // ws[0..63]=cnt, ws[64..127]=psum
}

// ---------------------------------------------------------------- prepack W -> bf16 WMMA-B tiles
// Layout: Wp[((s*4 + t)*32 + lane)*16 + j] = W[16t + (lane&15)][32s + 8*(lane>>4) + (j&7) + 16*(j>>3)]
__global__ __launch_bounds__(256) void prepack_w_k(const float* __restrict__ W,
                                                   __bf16* __restrict__ Wp) {
    int idx = blockIdx.x * 256 + threadIdx.x;   // 0 .. 131071
    int j = idx & 15;
    int L = (idx >> 4) & 31;
    int t = (idx >> 9) & 3;
    int s = idx >> 11;
    int e = 16 * t + (L & 15);
    int k = 32 * s + 8 * (L >> 4) + (j & 7) + 16 * (j >> 3);
    Wp[idx] = (__bf16)W[e * HDIM + k];
}

// 32-bit LDS byte offset of a __shared__ object (for the TDM descriptor)
__device__ static inline unsigned lds_offset_u32(const void* p) {
    return (unsigned)(unsigned long long)(__attribute__((address_space(3))) const void*)p;
}

// ---------------------------------------------------------------- main: GEMM + softmax + top-2
__global__ __launch_bounds__(256) void router_main_k(const float* __restrict__ x,
                                                     const __bf16* __restrict__ Wp,
                                                     float* __restrict__ out,
                                                     float* __restrict__ cnt,
                                                     float* __restrict__ psum) {
    __shared__ __attribute__((aligned(256))) __bf16 ldsWp[WP_ELEMS];   // 256 KB B staging
    __shared__ float lg[4][2][16][LROW];                               // ~33 KB epilogue

    const int tid   = threadIdx.x;
    const int lane  = tid & 31;
    const int wave  = tid >> 5;
    const int strip = wave >> 1;      // 4 row-strips per block
    const int khalf = wave & 1;       // K split across wave pairs -> 2x occupancy
    const int m     = lane & 15;
    const int half  = lane >> 4;
    const int r0    = blockIdx.x * 64 + strip * 16;

    // ---- TDM: pull the whole packed-W image (256 KB) into LDS with one descriptor.
    // Viewed as a 1 x 32768 tensor of 8-byte elements.
    if (wave == 0) {
        const unsigned long long ga = (unsigned long long)Wp;
        v4u g0;
        g0.x = 1u;                                      // count=1 (valid), user mode
        g0.y = lds_offset_u32(ldsWp);                   // lds_addr
        g0.z = (unsigned)(ga & 0xFFFFFFFFull);          // global_addr[31:0]
        g0.w = (unsigned)((ga >> 32) & 0x01FFFFFFull)   // global_addr[56:32]
             | (2u << 30);                              // type = 2 ("image")
        v8i g1;
        g1[0] = (int)(3u << 16);                        // data_size = 3 (8B), no multicast
        g1[1] = (int)0x80000000u;                       // tensor_dim0 = 32768 (lo16 in [31:16])
        g1[2] = (int)(1u << 16);                        // tensor_dim0 hi=0; tensor_dim1 = 1
        g1[3] = (int)0x80000000u;                       // tensor_dim1 hi=0; tile_dim0 = 32768
        g1[4] = 1;                                      // tile_dim1 = 1
        g1[5] = 32768;                                  // tensor_dim0_stride (lo32)
        g1[6] = 0;
        g1[7] = 0;
        v4i z4 = (v4i){0, 0, 0, 0};
        v8i z8 = (v8i){0, 0, 0, 0, 0, 0, 0, 0};
        __builtin_amdgcn_tensor_load_to_lds(g0, g1, z4, z4, z8, 0);
        __builtin_amdgcn_s_wait_tensorcnt(0);
    }
    __syncthreads();

    v8f acc0 = {}, acc1 = {}, acc2 = {}, acc3 = {};

    const float* arow = x + (size_t)(r0 + m) * HDIM;
    const int s0 = khalf * 32;        // 32 K-steps of 32 each = K range of 1024

    for (int ss = 0; ss < 32; ++ss) {
        const int s  = s0 + ss;
        const int kb = s * 32;
        // A tile 16x32 bf16 (streaming x: non-temporal)
        v4f a0 = __builtin_nontemporal_load((const v4f*)(arow + kb + 8 * half));
        v4f a1 = __builtin_nontemporal_load((const v4f*)(arow + kb + 8 * half + 4));
        v4f a2 = __builtin_nontemporal_load((const v4f*)(arow + kb + 16 + 8 * half));
        v4f a3 = __builtin_nontemporal_load((const v4f*)(arow + kb + 16 + 8 * half + 4));
        v16bf A;
#pragma unroll
        for (int j = 0; j < 4; ++j) {
            A[j]      = (__bf16)a0[j];
            A[4 + j]  = (__bf16)a1[j];
            A[8 + j]  = (__bf16)a2[j];
            A[12 + j] = (__bf16)a3[j];
        }
        // B tiles from LDS: 32B per lane per tile (2x ds_load_b128)
        const __bf16* bp = (const __bf16*)ldsWp + ((size_t)s * 128 + lane) * 16;
        v16bf B0 = *(const v16bf*)(bp);
        v16bf B1 = *(const v16bf*)(bp + 512);
        v16bf B2 = *(const v16bf*)(bp + 1024);
        v16bf B3 = *(const v16bf*)(bp + 1536);

        acc0 = __builtin_amdgcn_wmma_f32_16x16x32_bf16(false, A, false, B0, (short)0, acc0, false, false);
        acc1 = __builtin_amdgcn_wmma_f32_16x16x32_bf16(false, A, false, B1, (short)0, acc1, false, false);
        acc2 = __builtin_amdgcn_wmma_f32_16x16x32_bf16(false, A, false, B2, (short)0, acc2, false, false);
        acc3 = __builtin_amdgcn_wmma_f32_16x16x32_bf16(false, A, false, B3, (short)0, acc3, false, false);
    }

    // Spill partial logits to LDS. C/D layout: row = v + 8*half, col = m (+16 per tile)
    {
        float* Lw = &lg[strip][khalf][0][0];
#pragma unroll
        for (int v = 0; v < 8; ++v) {
            const int row = v + 8 * half;
            float* p = Lw + row * LROW + m;
            p[0]  = acc0[v];
            p[16] = acc1[v];
            p[32] = acc2[v];
            p[48] = acc3[v];
        }
    }
    __syncthreads();

    // Epilogue: one lane per token row (64 rows/block handled by 4 half-waves)
    if (khalf == 0 && lane < 16) {
        const int row = lane;
        float*       La = &lg[strip][0][row][0];
        const float* Lb = &lg[strip][1][row][0];

        float v1 = -3.0e38f, v2 = -3.0e38f;
        int   i1 = 0, i2 = 0;
        for (int e = 0; e < NEXP; ++e) {
            float l = La[e] + Lb[e];   // combine K-halves
            La[e] = l;
            if (l > v1)      { v2 = v1; i2 = i1; v1 = l; i1 = e; }
            else if (l > v2) { v2 = l;  i2 = e; }
        }
        // full softmax; v1 is the row max
        float sum = 0.0f;
        for (int e = 0; e < NEXP; ++e) sum += __expf(La[e] - v1);
        const float inv = 1.0f / sum;
        const int r = r0 + row;
        float* probs = out + PROB_OFF + (size_t)r * NEXP;
        for (int e = 0; e < NEXP; ++e) {
            float p = __expf(La[e] - v1) * inv;
            probs[e] = p;
            La[e] = p;                 // keep probs in LDS for column sums
        }
        // softmax over top-2 (descending order: v1 >= v2)
        const float w1 = 1.0f / (1.0f + __expf(v2 - v1));
        out[IDX_OFF + 2 * r + 0] = (float)i1;
        out[IDX_OFF + 2 * r + 1] = (float)i2;
        out[WT_OFF  + 2 * r + 0] = w1;
        out[WT_OFF  + 2 * r + 1] = 1.0f - w1;
        atomicAdd(&cnt[i1], 1.0f);
        atomicAdd(&cnt[i2], 1.0f);
    }
    __syncthreads();

    // per-expert prob sums: thread e sums probs over this block's 64 rows
    if (tid < NEXP) {
        float s = 0.0f;
        for (int st = 0; st < 4; ++st)
            for (int row = 0; row < 16; ++row)
                s += lg[st][0][row][tid];
        atomicAdd(&psum[tid], s);
    }
}

// ---------------------------------------------------------------- finalize: fraction + loss
__global__ __launch_bounds__(64) void finalize_k(const float* __restrict__ ws,
                                                 float* __restrict__ out) {
    __shared__ float red[NEXP];
    const int e = threadIdx.x;
    const float frac = ws[e]      * (1.0f / (float)NTOK);
    const float avg  = ws[64 + e] * (1.0f / (float)NTOK);
    out[FRAC_OFF + e] = frac;
    red[e] = frac * avg;
    __syncthreads();
    for (int off = 32; off > 0; off >>= 1) {
        if (e < off) red[e] += red[e + off];
        __syncthreads();
    }
    if (e == 0) out[LOSS_OFF] = (float)NEXP * red[0];
}

// ---------------------------------------------------------------- host launcher
extern "C" void kernel_launch(void* const* d_in, const int* in_sizes, int n_in,
                              void* d_out, int out_size, void* d_ws, size_t ws_size,
                              hipStream_t stream) {
    const float* x = (const float*)d_in[0];   // (16384, 2048) f32
    const float* W = (const float*)d_in[1];   // (64, 2048) f32
    float* out = (float*)d_out;
    float* ws  = (float*)d_ws;
    float*  cnt  = ws;                         // 64 floats
    float*  psum = ws + 64;                    // 64 floats
    __bf16* Wp   = (__bf16*)(ws + 128);        // 131072 bf16 = 256 KB, 512B-aligned

    zero_ws_k<<<1, 128, 0, stream>>>(ws);
    prepack_w_k<<<WP_ELEMS / 256, 256, 0, stream>>>(W, Wp);
    router_main_k<<<NTOK / 64, 256, 0, stream>>>(x, Wp, out, cnt, psum);
    finalize_k<<<1, 64, 0, stream>>>(ws, out);
}
